// QuantumQLSTM_65481071403446
// MI455X (gfx1250) — compile-verified, matrix operand order checked by reading, not run on hardware
//
#include <hip/hip_runtime.h>
#include <hip/hip_bf16.h>
#include <math.h>

typedef __attribute__((ext_vector_type(16))) _Float16 v16h;
typedef __attribute__((ext_vector_type(8)))  float    v8f;

#define NQ     8
#define INDIM  256
#define DTOT   264      // 256 + 8
#define HIDN   8
#define BB     512
#define TT     512
#define MROWS  (BB*TT)  // 262144 rows of the big GEMM

// ---------------------------------------------------------------------------
// Kernel 1: pack the 32x256 gate-weight matrix (x-columns only) into f16
// B-matrix fragments for v_wmma_f32_16x16x32_f16.
// B is KxN = 32x16 per fragment; per ISA 7.12.2 (16-bit operand K striping):
//   lane = {h=lane/16, c=lane%16}, half i = {v=i/2, j=i&1}
//   Klocal = (v<4 ? 2v+8h : 16+2(v-4)+8h) + j,  N = c
// Fragment storage: wpack[((kc*2 + nt)*32 + lane)*16 + i], kc=K/32, nt=N/16.
// ---------------------------------------------------------------------------
__global__ void __launch_bounds__(256)
pack_w(const float* __restrict__ Wf, const float* __restrict__ Wi,
       const float* __restrict__ Wg, const float* __restrict__ Wo,
       _Float16* __restrict__ wpack) {
  int idx = blockIdx.x * blockDim.x + threadIdx.x;
  if (idx >= 8 * 2 * 32 * 16) return;
  int i    = idx & 15;
  int lane = (idx >> 4) & 31;
  int nt   = (idx >> 9) & 1;
  int kc   = idx >> 10;
  int v = i >> 1, j = i & 1;
  int h = lane >> 4, c = lane & 15;
  int kl = (v < 4 ? 2 * v + 8 * h : 16 + 2 * (v - 4) + 8 * h) + j;
  int K = kc * 32 + kl;          // input-feature index d (0..255)
  int N = nt * 16 + c;           // flattened gate*8+qubit (0..31)
  int g = N >> 3, q = N & 7;
  const float* Wsrc = (g == 0) ? Wf : (g == 1) ? Wi : (g == 2) ? Wg : Wo;
  wpack[idx] = (_Float16)Wsrc[q * DTOT + K];
}

// ---------------------------------------------------------------------------
// Kernel 2: pre[m][n] = sum_d x[m][d] * Wcat[n][d],  m = b*T+t, n = gate*8+q.
// One wave per 16(M)x32(N) tile: two f32 accumulators, 8 K-chunks of 32,
// => 16 v_wmma_f32_16x16x32_f16 per wave. Memory-bound: x read once (256MB).
// ---------------------------------------------------------------------------
__global__ void __launch_bounds__(256)
gemm_pre(const float* __restrict__ x, const _Float16* __restrict__ wpack,
         float* __restrict__ pre) {
  const int wave = threadIdx.x >> 5;
  const int lane = threadIdx.x & 31;
  const int tile = blockIdx.x * 8 + wave;
  const int m0 = tile * 16;
  const int h = lane >> 4, r = lane & 15;

  const float* __restrict__ arow = x + (size_t)(m0 + r) * INDIM;
  v8f acc0 = {};
  v8f acc1 = {};

#pragma unroll
  for (int kc = 0; kc < 8; ++kc) {
    // A fragment: row M=r (this lane), K values per ISA striping.
    // v0..v3 -> K = kc*32 + 8h + 0..7 ; v4..v7 -> K = kc*32 + 16 + 8h + 0..7
    const int kb = kc * 32 + 8 * h;
    float4 p0 = *(const float4*)(arow + kb);
    float4 p1 = *(const float4*)(arow + kb + 4);
    float4 p2 = *(const float4*)(arow + kb + 16);
    float4 p3 = *(const float4*)(arow + kb + 20);
    v16h a;
    a[0]  = (_Float16)p0.x; a[1]  = (_Float16)p0.y;
    a[2]  = (_Float16)p0.z; a[3]  = (_Float16)p0.w;
    a[4]  = (_Float16)p1.x; a[5]  = (_Float16)p1.y;
    a[6]  = (_Float16)p1.z; a[7]  = (_Float16)p1.w;
    a[8]  = (_Float16)p2.x; a[9]  = (_Float16)p2.y;
    a[10] = (_Float16)p2.z; a[11] = (_Float16)p2.w;
    a[12] = (_Float16)p3.x; a[13] = (_Float16)p3.y;
    a[14] = (_Float16)p3.z; a[15] = (_Float16)p3.w;

    const v16h b0 = *(const v16h*)(wpack + (size_t)((kc * 2 + 0) * 32 + lane) * 16);
    const v16h b1 = *(const v16h*)(wpack + (size_t)((kc * 2 + 1) * 32 + lane) * 16);

    acc0 = __builtin_amdgcn_wmma_f32_16x16x32_f16(false, a, false, b0,
                                                  (short)0, acc0, false, false);
    acc1 = __builtin_amdgcn_wmma_f32_16x16x32_f16(false, a, false, b1,
                                                  (short)0, acc1, false, false);
  }

  // f32 C/D layout: VGPR d -> M = m0 + d + 8h, N = lane%16 (+16 for acc1)
#pragma unroll
  for (int d = 0; d < 8; ++d) {
    const size_t row = (size_t)(m0 + d + 8 * h);
    pre[row * 32 + r]      = acc0[d];
    pre[row * 32 + 16 + r] = acc1[d];
  }
}

// ---------------------------------------------------------------------------
// Kernel 3: batch-parallel recurrence. One wave per batch row; lane n = g*8+q.
// ang_n(t) = pre[b][t][n] + bias_n + theta_n + sum_j h_j * W_n[256+j]
// Quantum circuit collapses to m[q] = prod_{w in S_q} cos(ang_{g,w}),
//   S_0 = {1..7}, S_q = {0..q} for q>=1.
// ---------------------------------------------------------------------------
__global__ void __launch_bounds__(256)
recur(const float* __restrict__ pre,
      const float* __restrict__ Wf, const float* __restrict__ bf,
      const float* __restrict__ Wi, const float* __restrict__ bi,
      const float* __restrict__ Wg, const float* __restrict__ bg,
      const float* __restrict__ Wo, const float* __restrict__ bo,
      const float* __restrict__ thetas,
      float* __restrict__ out) {
  const int lane = threadIdx.x & 31;
  const int wave = threadIdx.x >> 5;
  const int b = blockIdx.x * 8 + wave;
  const int g = lane >> 3, q = lane & 7;

  const float* Wsrc = (g == 0) ? Wf : (g == 1) ? Wi : (g == 2) ? Wg : Wo;
  const float* bsrc = (g == 0) ? bf : (g == 1) ? bi : (g == 2) ? bg : bo;

  float wh[8];
#pragma unroll
  for (int j = 0; j < 8; ++j) wh[j] = Wsrc[q * DTOT + INDIM + j];
  const float bconst = bsrc[q] + thetas[g * 8 + q];

  // lanes {q, q+8, q+16, q+24} all carry hx[q]/cx[q] (kept uniform below)
  float h = 0.0f, c = 0.0f;

  const size_t base = (size_t)b * TT;
  float nxt = pre[base * 32 + lane];   // 1-step software prefetch

  for (int t = 0; t < TT; ++t) {
    float ang = nxt + bconst;
    if (t + 1 < TT) nxt = pre[(base + (size_t)(t + 1)) * 32 + lane];

#pragma unroll
    for (int j = 0; j < 8; ++j)
      ang += __shfl(h, j, 32) * wh[j];   // lane j holds hx[j]

    const float cw = cosf(ang);

    // gather the 8 cosines of this gate's group
    const int gb = lane & 24;
    float cv[8];
#pragma unroll
    for (int w = 0; w < 8; ++w) cv[w] = __shfl(cw, gb + w, 32);

    // m[q] = product over S_q
    float m = 1.0f;
#pragma unroll
    for (int w = 0; w < 8; ++w) {
      const bool incl = (q == 0) ? (w >= 1) : (w <= q);
      m = incl ? m * cv[w] : m;
    }

    // redistribute by qubit: every lane fetches gate values for its q
    const float mf = __shfl(m, q, 32);
    const float mi = __shfl(m, 8 + q, 32);
    const float mg = __shfl(m, 16 + q, 32);
    const float mo = __shfl(m, 24 + q, 32);

    const float fg = 1.0f / (1.0f + expf(-mf));
    const float ig = 1.0f / (1.0f + expf(-mi));
    const float gg = tanhf(mg);
    const float og = 1.0f / (1.0f + expf(-mo));

    c = fg * c + ig * gg;
    h = og * tanhf(c);

    if (lane < 8) out[(base + (size_t)t) * 8 + q] = h;
  }

  if (lane < 8) {
    out[(size_t)MROWS * 8 + (size_t)b * 8 + q] = h;                 // hx
    out[(size_t)MROWS * 8 + (size_t)BB * 8 + (size_t)b * 8 + q] = c; // cx
  }
}

// ---------------------------------------------------------------------------
extern "C" void kernel_launch(void* const* d_in, const int* in_sizes, int n_in,
                              void* d_out, int out_size, void* d_ws, size_t ws_size,
                              hipStream_t stream) {
  (void)in_sizes; (void)n_in; (void)out_size; (void)ws_size;
  const float* x      = (const float*)d_in[0];
  const float* Wf     = (const float*)d_in[1];
  const float* bf     = (const float*)d_in[2];
  const float* Wi     = (const float*)d_in[3];
  const float* bi     = (const float*)d_in[4];
  const float* Wg     = (const float*)d_in[5];
  const float* bg     = (const float*)d_in[6];
  const float* Wo     = (const float*)d_in[7];
  const float* bo     = (const float*)d_in[8];
  const float* thetas = (const float*)d_in[9];

  _Float16* wpack = (_Float16*)d_ws;                    // 16 KB
  float*    pre   = (float*)((char*)d_ws + 16384);      // 32 MB: [M][32]

  pack_w<<<32, 256, 0, stream>>>(Wf, Wi, Wg, Wo, wpack);
  gemm_pre<<<MROWS / 16 / 8, 256, 0, stream>>>(x, wpack, pre);  // 2048 blocks
  recur<<<BB / 8, 256, 0, stream>>>(pre, Wf, bf, Wi, bi, Wg, bg, Wo, bo,
                                    thetas, (float*)d_out);
}
